// LearnCut_58291296141642
// MI455X (gfx1250) — matrix operand order repaired and checked by reading, not compiled, not run on hardware
//
#include <hip/hip_runtime.h>
#include <hip/hip_bf16.h>
#include <math.h>

// ---------------- problem constants ----------------
constexpr int Bc = 2, Mc = 2048, Nc = 16384, Cc = 256, HIDc = 256;
constexpr int Kc = 16, NCLUSc = 64, KNNc = 16, NFREQc = 12;
constexpr float TAUc = 0.1f;
constexpr int BM  = Bc * Mc;        // 4096 rows
constexpr int BMK = BM * Kc;        // 65536 rows
constexpr int PADF = 96;            // 72 Fourier features zero-padded to 96 (multiple of 32)

// ---------------- workspace layout (floats) ----------------
constexpr size_t O_POSE_S = 0;
constexpr size_t O_POSH   = O_POSE_S + (size_t)BM * PADF;
constexpr size_t O_POSEMD = O_POSH   + (size_t)BM * 128;
constexpr size_t O_QIN    = O_POSEMD + (size_t)BM * Cc;
constexpr size_t O_Q      = O_QIN    + (size_t)BM * Cc;
constexpr size_t O_POSE_R = O_Q      + (size_t)BM * Cc;
constexpr size_t O_NEIH   = O_POSE_R + (size_t)BMK * PADF;
constexpr size_t O_KV     = O_NEIH   + (size_t)BMK * 128;
constexpr size_t O_KF     = O_KV     + (size_t)BMK * Cc;
constexpr size_t O_V      = O_KF     + (size_t)BMK * Cc;
constexpr size_t O_ATTN   = O_V      + (size_t)BMK * Cc;
constexpr size_t O_X      = O_ATTN   + (size_t)BM * Cc;
constexpr size_t O_HPRE   = O_X      + (size_t)BM * Cc;
constexpr size_t O_HBN    = O_HPRE   + (size_t)BM * Cc;
constexpr size_t O_FPRE   = O_HBN    + (size_t)BM * Cc;
constexpr size_t O_ADJ    = O_FPRE   + (size_t)BM * Cc;
constexpr size_t O_DINV   = O_ADJ    + (size_t)Bc * Mc * Mc;
constexpr size_t O_HG     = O_DINV   + (size_t)BM;
constexpr size_t O_CUT1   = O_HG     + (size_t)BM * Cc;
constexpr size_t O_SLOG   = O_CUT1   + (size_t)BM * Cc;
constexpr size_t O_S      = O_SLOG   + (size_t)BM * NCLUSc;
constexpr size_t O_CTSUM  = O_S      + (size_t)BM * NCLUSc;
constexpr size_t O_SSUM   = O_CTSUM  + (size_t)Bc * NCLUSc * 3;
constexpr size_t O_MEAN1  = O_SSUM   + (size_t)Bc * NCLUSc;
constexpr size_t O_VAR1   = O_MEAN1  + HIDc;
constexpr size_t O_MEAN2  = O_VAR1   + HIDc;
constexpr size_t O_VAR2   = O_MEAN2  + Cc;
// packed-f16 B-operand arena (in halves), 32B-aligned start
constexpr size_t O_PACKF  = ((O_VAR2 + Cc + 15) & ~(size_t)15);

// packed B sizes (halves): (Kdim/32) * (Ncols/16) * 512
constexpr size_t pkSize(int Kd, int Nc_) { return (size_t)(Kd / 32) * (Nc_ / 16) * 512; }
constexpr size_t PK_POSW1  = 0;
constexpr size_t PK_POSW2  = PK_POSW1  + pkSize(PADF, 128);
constexpr size_t PK_NEIW1  = PK_POSW2  + pkSize(128, Cc);
constexpr size_t PK_NEIW2  = PK_NEIW1  + pkSize(PADF, 128);
constexpr size_t PK_QW     = PK_NEIW2  + pkSize(128, Cc);
constexpr size_t PK_KW     = PK_QW     + pkSize(Cc, HIDc);
constexpr size_t PK_VW     = PK_KW     + pkSize(Cc, HIDc);
constexpr size_t PK_PROJW1 = PK_VW     + pkSize(Cc, HIDc);
constexpr size_t PK_PROJW2 = PK_PROJW1 + pkSize(Cc, HIDc);
constexpr size_t PK_CUTW1  = PK_PROJW2 + pkSize(HIDc, Cc);
constexpr size_t PK_CUTW2  = PK_CUTW1  + pkSize(Cc, HIDc);
constexpr size_t PK_SPF    = PK_CUTW2  + pkSize(HIDc, NCLUSc);   // 2 batches follow

// ---------------- WMMA types ----------------
typedef __attribute__((ext_vector_type(16))) _Float16 v16h;
typedef __attribute__((ext_vector_type(8)))  float    v8f;
union HalfVec { v16h v; _Float16 h[16]; };

enum { EPI_NONE = 0, EPI_BIAS = 1, EPI_GELU = 2, EPI_RELU = 3, EPI_LN = 4 };

__device__ __forceinline__ void cvt4(_Float16* d, float4 s) {
  d[0] = (_Float16)s.x; d[1] = (_Float16)s.y;
  d[2] = (_Float16)s.z; d[3] = (_Float16)s.w;
}

// ----------------------------------------------------------------------
// Pack a KxN f32 matrix (row-major) into WMMA B-fragment order, f16.
// Element (kc, tn, lane, j) <- W[kc*32 + (lane>=16?8:0) + (j&7) + (j>=8?16:0), tn*16 + (lane&15)]
// Rows >= KdimSrc are zero (used to pad 72 -> 96).
// ----------------------------------------------------------------------
__global__ void pack_b_kernel(const float* __restrict__ W, _Float16* __restrict__ out,
                              int KdimSrc, int Kdim, int Ncols) {
  size_t total = (size_t)(Kdim / 32) * (Ncols / 16) * 512;
  size_t tid = (size_t)blockIdx.x * blockDim.x + threadIdx.x;
  if (tid >= total) return;
  int j = (int)(tid & 15);
  int l = (int)((tid >> 4) & 31);
  size_t t = tid >> 9;
  int ntiles = Ncols >> 4;
  int tn = (int)(t % ntiles);
  int kc = (int)(t / ntiles);
  int k   = kc * 32 + (l >= 16 ? 8 : 0) + (j & 7) + ((j >> 3) << 4);
  int col = tn * 16 + (l & 15);
  float v = (k < KdimSrc) ? W[(size_t)k * Ncols + col] : 0.f;
  out[tid] = (_Float16)v;
}

// ======================================================================
// WMMA GEMM: Out[Mrows x Ncols] = epi(A[Mrows x Kdim] @ B) with B pre-packed
// f16 fragments. Block = 128 threads (4 waves) covering a 16-row strip of the
// FULL width (Ncols == TPW*64), so LayerNorm is block-local. Kdim % 32 == 0.
// ======================================================================
template <int EPI, int TPW>
__global__ __launch_bounds__(128)
void gemm_wmma(const float* __restrict__ A, const _Float16* __restrict__ Bp,
               const float* __restrict__ bias, const float* __restrict__ lng,
               const float* __restrict__ lnb, float* __restrict__ Out,
               int Mrows, int Kdim, int Ncols) {
  const int lane    = threadIdx.x & 31;
  const int wave    = threadIdx.x >> 5;
  const int row0    = blockIdx.x << 4;
  const int tile0   = wave * TPW;          // first 16-col tile of this wave
  const int ntiles  = Ncols >> 4;
  const int halfSel = lane >> 4;
  const int lmod    = lane & 15;

  v8f acc[TPW] = {};

  const float* ap = A + (size_t)(row0 + lmod) * Kdim + (halfSel ? 8 : 0);
  for (int k0 = 0; k0 < Kdim; k0 += 32) {
    __builtin_prefetch(ap + k0 + 32, 0, 1);          // global_prefetch_b8
    HalfVec af;
    {
      const float4* p = (const float4*)(ap + k0);
      float4 x0 = p[0], x1 = p[1];                   // k: +0..7
      const float4* q = (const float4*)(ap + k0 + 16);
      float4 x2 = q[0], x3 = q[1];                   // k: +16..23
      cvt4(&af.h[0], x0); cvt4(&af.h[4], x1);
      cvt4(&af.h[8], x2); cvt4(&af.h[12], x3);
    }
    const size_t kbase = ((size_t)(k0 >> 5) * ntiles) << 9;   // *512
#pragma unroll
    for (int t = 0; t < TPW; ++t) {
      const v16h bf = *(const v16h*)(Bp + kbase + ((size_t)(tile0 + t) << 9)
                                        + ((size_t)lane << 4));
      acc[t] = __builtin_amdgcn_wmma_f32_16x16x32_f16(
          false, af.v, false, bf, (short)0, acc[t], false, false);
    }
  }

  __shared__ float lsum[16];
  __shared__ float lssq[16];
  if constexpr (EPI == EPI_LN) {
    if (threadIdx.x < 16) { lsum[threadIdx.x] = 0.f; lssq[threadIdx.x] = 0.f; }
    __syncthreads();
#pragma unroll
    for (int t = 0; t < TPW; ++t) {
      const int col = ((tile0 + t) << 4) + lmod;
      const float bv = bias ? bias[col] : 0.f;
#pragma unroll
      for (int i = 0; i < 8; ++i) {
        int lr = halfSel ? i + 8 : i;
        float v = acc[t][i] + bv;
        atomicAdd(&lsum[lr], v);                     // ds_add_f32
        atomicAdd(&lssq[lr], v * v);
      }
    }
    __syncthreads();
  }

#pragma unroll
  for (int t = 0; t < TPW; ++t) {
    const int col = ((tile0 + t) << 4) + lmod;
    const float bv = (EPI != EPI_NONE && bias) ? bias[col] : 0.f;
#pragma unroll
    for (int i = 0; i < 8; ++i) {
      int lr = halfSel ? i + 8 : i;
      int r  = row0 + lr;
      float v = acc[t][i] + bv;
      if constexpr (EPI == EPI_GELU) {
        float x3 = v * v * v;
        v = 0.5f * v * (1.f + tanhf(0.7978845608f * (v + 0.044715f * x3)));
      } else if constexpr (EPI == EPI_RELU) {
        v = fmaxf(v, 0.f);
      } else if constexpr (EPI == EPI_LN) {
        float mean = lsum[lr] / (float)Ncols;
        float var  = lssq[lr] / (float)Ncols - mean * mean;
        v = (v - mean) * rsqrtf(var + 1e-5f) * lng[col] + lnb[col];
      }
      Out[(size_t)r * Ncols + col] = v;
    }
  }
}

// ---------------- block reduce (256 threads) ----------------
__device__ __forceinline__ float blockReduceSum256(float v, float* sh) {
  int c = threadIdx.x;
  sh[c] = v; __syncthreads();
#pragma unroll
  for (int s = 128; s > 0; s >>= 1) {
    if (c < s) sh[c] += sh[c + s];
    __syncthreads();
  }
  float r = sh[0];
  __syncthreads();
  return r;
}

// ---------------- Fourier positional encoding (96-wide, zero-padded) -------
__global__ void pose_sp_kernel(const float* __restrict__ xyz, float* __restrict__ out) {
  size_t tid = (size_t)blockIdx.x * blockDim.x + threadIdx.x;
  if (tid >= (size_t)BM * PADF) return;
  int e = (int)(tid % PADF); size_t row = tid / PADF;
  if (e >= 72) { out[tid] = 0.f; return; }
  int cidx = e / 24, j = e % 24;
  int f = (j < 12) ? j : j - 12;
  float div = powf(100.f, (float)f / (float)NFREQc);
  float ang = xyz[row * 3 + cidx] * 100.f / div;
  out[tid] = (j < 12) ? sinf(ang) : cosf(ang);
}

__global__ void pose_rel_kernel(const float* __restrict__ sp_xyz,
                                const float* __restrict__ p_xyz,
                                const int* __restrict__ topk,
                                float* __restrict__ out) {
  size_t tid = (size_t)blockIdx.x * blockDim.x + threadIdx.x;
  if (tid >= (size_t)BMK * PADF) return;
  int e = (int)(tid % PADF); size_t row = tid / PADF;     // (b,m,k)
  if (e >= 72) { out[tid] = 0.f; return; }
  size_t bm = row / Kc; int b = (int)(bm / Mc);
  int cidx = e / 24, j = e % 24;
  int f = (j < 12) ? j : j - 12;
  int idx = topk[row];
  float x = p_xyz[((size_t)b * Nc + idx) * 3 + cidx] - sp_xyz[bm * 3 + cidx];
  float ang = x * 100.f / powf(100.f, (float)f / (float)NFREQc);
  out[tid] = (j < 12) ? sinf(ang) : cosf(ang);
}

// kv[row,c] += gathered p_fea  (neigh_fea + neigh_pos in place)
__global__ void kv_add_kernel(const float* __restrict__ p_fea,
                              const int* __restrict__ topk,
                              float* __restrict__ kv) {
  size_t tid = (size_t)blockIdx.x * blockDim.x + threadIdx.x;
  if (tid >= (size_t)BMK * Cc) return;
  int c = (int)(tid % Cc); size_t row = tid / Cc;
  int b = (int)(row / ((size_t)Mc * Kc));
  int idx = topk[row];
  kv[tid] += p_fea[((size_t)b * Nc + idx) * Cc + c];
}

__global__ void add_kernel(const float* __restrict__ a, const float* __restrict__ b,
                           float* __restrict__ out, size_t n) {
  size_t tid = (size_t)blockIdx.x * blockDim.x + threadIdx.x;
  if (tid < n) out[tid] = a[tid] + b[tid];
}

// ---------------- attention over K=16 neighbors ----------------
__global__ __launch_bounds__(256)
void attn_kernel(const float* __restrict__ q, const float* __restrict__ kf,
                 const float* __restrict__ v, const int* __restrict__ mask,
                 float* __restrict__ attn_fea) {
  __shared__ float sh[256];
  __shared__ float lg[Kc];
  __shared__ float aw[Kc];
  int c = threadIdx.x;
  size_t row = blockIdx.x;                       // b*M + m
  float qc = q[row * Cc + c];
  for (int k = 0; k < Kc; ++k) {
    float p = qc * kf[(row * Kc + k) * Cc + c];
    float s = blockReduceSum256(p, sh);
    if (c == 0) lg[k] = s * (1.0f / 16.0f);      // / sqrt(C)
  }
  __syncthreads();
  if (c == 0) {
    bool any = false;
    for (int k = 0; k < Kc; ++k) any = any || (mask[row * Kc + k] != 0);
    float mx = -1e30f;
    for (int k = 0; k < Kc; ++k) {
      float l = lg[k];
      if (any && mask[row * Kc + k] == 0) l = -1e9f;
      lg[k] = l; mx = fmaxf(mx, l);
    }
    float s = 0.f;
    for (int k = 0; k < Kc; ++k) { float e = __expf(lg[k] - mx); aw[k] = e; s += e; }
    float inv = 1.0f / s;
    for (int k = 0; k < Kc; ++k) aw[k] *= inv;
  }
  __syncthreads();
  float a = 0.f;
  for (int k = 0; k < Kc; ++k) a += aw[k] * v[(row * Kc + k) * Cc + c];
  attn_fea[row * Cc + c] = a;
}

// x = LN(a + b; g, be)
__global__ __launch_bounds__(256)
void add_ln_kernel(const float* __restrict__ a, const float* __restrict__ b,
                   const float* __restrict__ g, const float* __restrict__ be,
                   float* __restrict__ out) {
  __shared__ float sh[256];
  int c = threadIdx.x; size_t row = blockIdx.x;
  float v = a[row * Cc + c] + b[row * Cc + c];
  float m = blockReduceSum256(v, sh) * (1.0f / Cc);
  float d = v - m;
  float var = blockReduceSum256(d * d, sh) * (1.0f / Cc);
  out[row * Cc + c] = d * rsqrtf(var + 1e-5f) * g[c] + be[c];
}

// ---------------- batchnorm: stats + apply(relu) ----------------
__global__ __launch_bounds__(256)
void bn_stats_kernel(const float* __restrict__ x, float* __restrict__ mean,
                     float* __restrict__ var, int rows, int cols) {
  __shared__ float sh[256];
  int ch = blockIdx.x;
  float s = 0.f, ss = 0.f;
  for (int r = threadIdx.x; r < rows; r += 256) {
    float v = x[(size_t)r * cols + ch];
    s += v; ss += v * v;
  }
  s  = blockReduceSum256(s, sh);
  ss = blockReduceSum256(ss, sh);
  if (threadIdx.x == 0) {
    float m = s / rows;
    mean[ch] = m;
    var[ch]  = ss / rows - m * m;
  }
}

__global__ void bn_relu_kernel(const float* __restrict__ x, const float* __restrict__ mean,
                               const float* __restrict__ var, const float* __restrict__ g,
                               const float* __restrict__ b, float* __restrict__ out,
                               size_t n, int cols) {
  size_t tid = (size_t)blockIdx.x * blockDim.x + threadIdx.x;
  if (tid >= n) return;
  int ch = (int)(tid % cols);
  float v = (x[tid] - mean[ch]) * rsqrtf(var[ch] + 1e-5f) * g[ch] + b[ch];
  out[tid] = fmaxf(v, 0.f);
}

// loss += mean_b sum_m ||l2n(leaky(a)) - l2n(f)|| / C
__global__ __launch_bounds__(256)
void norm_loss_kernel(const float* __restrict__ af, const float* __restrict__ sf,
                      float* __restrict__ loss, float slope) {
  __shared__ float sh[256];
  int c = threadIdx.x; size_t row = blockIdx.x;
  float a = af[row * Cc + c]; a = a > 0.f ? a : slope * a;
  float f = sf[row * Cc + c];
  float na = fmaxf(sqrtf(blockReduceSum256(a * a, sh)), 1e-12f);
  float nf = fmaxf(sqrtf(blockReduceSum256(f * f, sh)), 1e-12f);
  float d = a / na - f / nf;
  float dn = sqrtf(blockReduceSum256(d * d, sh));
  if (c == 0) atomicAdd(loss, dn * (1.0f / ((float)Bc * (float)Cc)));
}

// ---------------- min-cut branch ----------------
__global__ __launch_bounds__(128)
void knn_kernel(const float* __restrict__ xyz, float* __restrict__ Amat) {
  int tid = blockIdx.x * 128 + threadIdx.x;
  if (tid >= BM) return;
  int b = tid / Mc, m = tid % Mc;
  const float* xb = xyz + (size_t)b * Mc * 3;
  float px = xb[m * 3 + 0], py = xb[m * 3 + 1], pz = xb[m * 3 + 2];
  float bd[KNNc]; int bi[KNNc];
#pragma unroll
  for (int j = 0; j < KNNc; ++j) { bd[j] = 3.4e38f; bi[j] = 0; }
  for (int n = 0; n < Mc; ++n) {
    float dx = px - xb[n * 3 + 0];
    float dy = py - xb[n * 3 + 1];
    float dz = pz - xb[n * 3 + 2];
    float d2 = dx * dx + dy * dy + dz * dz + (n == m ? 1e10f : 0.f);
    if (d2 < bd[KNNc - 1]) {
      int j = KNNc - 1;
      while (j > 0 && bd[j - 1] > d2) { bd[j] = bd[j - 1]; bi[j] = bi[j - 1]; --j; }
      bd[j] = d2; bi[j] = n;
    }
  }
  float* Ab = Amat + (size_t)b * Mc * Mc;
  for (int j = 0; j < KNNc; ++j) {
    float w = __expf(-bd[j]);                    // exp(negd), w > 0
    int wi = __float_as_int(w);                  // monotone for w >= 0
    atomicMax((int*)&Ab[(size_t)m * Mc + bi[j]], wi);   // A = max(A, A^T)
    atomicMax((int*)&Ab[(size_t)bi[j] * Mc + m], wi);
  }
}

__global__ __launch_bounds__(256)
void degree_kernel(const float* __restrict__ Amat, float* __restrict__ dinv) {
  __shared__ float sh[256];
  size_t row = blockIdx.x;                       // b*M + i
  int b = (int)(row / Mc);
  const float* Ar = Amat + (size_t)b * Mc * Mc + (row % Mc) * (size_t)Mc;
  float s = 0.f;
  for (int n = threadIdx.x; n < Mc; n += 256) s += Ar[n];
  s = blockReduceSum256(s, sh);
  if (threadIdx.x == 0) dinv[row] = rsqrtf(s + 1e-4f);
}

__global__ void scale_adj_kernel(float* __restrict__ Amat, const float* __restrict__ dinv) {
  size_t tid = (size_t)blockIdx.x * blockDim.x + threadIdx.x;
  if (tid >= (size_t)Bc * Mc * Mc) return;
  size_t j  = tid % Mc;
  size_t bi = tid / Mc;                          // b*M + i
  int b = (int)(bi / Mc);
  Amat[tid] *= dinv[bi] * dinv[(size_t)b * Mc + j];
}

__global__ __launch_bounds__(64)
void cut_softmax_kernel(const float* __restrict__ logits, float* __restrict__ s) {
  __shared__ float sh[64];
  int c = threadIdx.x; size_t row = blockIdx.x;
  float z = logits[row * NCLUSc + c] * (1.0f / TAUc);
  sh[c] = z; __syncthreads();
#pragma unroll
  for (int st = 32; st > 0; st >>= 1) { if (c < st) sh[c] = fmaxf(sh[c], sh[c + st]); __syncthreads(); }
  float mx = sh[0]; __syncthreads();
  float e = __expf(z - mx);
  sh[c] = e; __syncthreads();
#pragma unroll
  for (int st = 32; st > 0; st >>= 1) { if (c < st) sh[c] += sh[c + st]; __syncthreads(); }
  s[row * NCLUSc + c] = e / sh[0];
}

__global__ void cts_accum_kernel(const float* __restrict__ s, const float* __restrict__ xyz,
                                 float* __restrict__ ctsum, float* __restrict__ ssum) {
  int tid = blockIdx.x * 256 + threadIdx.x;
  if (tid >= BM) return;
  int b = tid / Mc;
  float x = xyz[(size_t)tid * 3 + 0], y = xyz[(size_t)tid * 3 + 1], z = xyz[(size_t)tid * 3 + 2];
  for (int k = 0; k < NCLUSc; ++k) {
    float sk = s[(size_t)tid * NCLUSc + k];
    size_t o = (size_t)b * NCLUSc + k;
    atomicAdd(&ctsum[o * 3 + 0], sk * x);
    atomicAdd(&ctsum[o * 3 + 1], sk * y);
    atomicAdd(&ctsum[o * 3 + 2], sk * z);
    atomicAdd(&ssum[o], sk);
  }
}

__global__ void cut_loss_kernel(const float* __restrict__ s, const float* __restrict__ xyz,
                                const float* __restrict__ ctsum, const float* __restrict__ ssum,
                                float* __restrict__ loss) {
  int tid = blockIdx.x * 256 + threadIdx.x;
  if (tid >= BM) return;
  int b = tid / Mc;
  float x = xyz[(size_t)tid * 3 + 0], y = xyz[(size_t)tid * 3 + 1], z = xyz[(size_t)tid * 3 + 2];
  float acc = 0.f;
  for (int k = 0; k < NCLUSc; ++k) {
    size_t o = (size_t)b * NCLUSc + k;
    float inv = 1.0f / (ssum[o] + 1e-6f);
    float dx = x - ctsum[o * 3 + 0] * inv;
    float dy = y - ctsum[o * 3 + 1] * inv;
    float dz = z - ctsum[o * 3 + 2] * inv;
    acc += s[(size_t)tid * NCLUSc + k] * (dx * dx + dy * dy + dz * dz);
  }
  atomicAdd(loss, acc * (1.0f / ((float)Bc * (float)Mc)));
}

// ======================================================================
extern "C" void kernel_launch(void* const* d_in, const int* in_sizes, int n_in,
                              void* d_out, int out_size, void* d_ws, size_t ws_size,
                              hipStream_t stream) {
  (void)in_sizes; (void)n_in; (void)out_size; (void)ws_size;
  const float* sp_fea = (const float*)d_in[0];
  const float* sp_xyz = (const float*)d_in[1];
  const float* p_fea  = (const float*)d_in[2];
  const float* p_xyz  = (const float*)d_in[3];
  const int*   topk   = (const int*)d_in[4];
  const int*   mask   = (const int*)d_in[5];
  int pi = 6;
  const float* pos_w1 = (const float*)d_in[pi++]; const float* pos_b1 = (const float*)d_in[pi++];
  const float* pos_w2 = (const float*)d_in[pi++]; const float* pos_b2 = (const float*)d_in[pi++];
  const float* pos_ln_g = (const float*)d_in[pi++]; const float* pos_ln_b = (const float*)d_in[pi++];
  const float* nei_w1 = (const float*)d_in[pi++]; const float* nei_b1 = (const float*)d_in[pi++];
  const float* nei_w2 = (const float*)d_in[pi++]; const float* nei_b2 = (const float*)d_in[pi++];
  const float* nei_ln_g = (const float*)d_in[pi++]; const float* nei_ln_b = (const float*)d_in[pi++];
  const float* q_w = (const float*)d_in[pi++]; const float* q_b = (const float*)d_in[pi++];
  const float* k_w = (const float*)d_in[pi++]; const float* k_b = (const float*)d_in[pi++];
  const float* v_w = (const float*)d_in[pi++]; const float* v_b = (const float*)d_in[pi++];
  const float* norm1_g = (const float*)d_in[pi++]; const float* norm1_b = (const float*)d_in[pi++];
  const float* norm2_g = (const float*)d_in[pi++]; const float* norm2_b = (const float*)d_in[pi++];
  const float* proj_w1 = (const float*)d_in[pi++]; const float* proj_b1 = (const float*)d_in[pi++];
  const float* proj_bn1_g = (const float*)d_in[pi++]; const float* proj_bn1_b = (const float*)d_in[pi++];
  const float* proj_w2 = (const float*)d_in[pi++];
  const float* proj_bn2_g = (const float*)d_in[pi++]; const float* proj_bn2_b = (const float*)d_in[pi++];
  const float* cut_w1 = (const float*)d_in[pi++]; const float* cut_b1 = (const float*)d_in[pi++];
  const float* cut_w2 = (const float*)d_in[pi++]; const float* cut_b2 = (const float*)d_in[pi++];

  float* W = (float*)d_ws;
  _Float16* HP = (_Float16*)(W + O_PACKF);      // packed-f16 arena
  float* out = (float*)d_out;
  float* spf_new = out;                         // (B,M,C)
  float* lossL = out + (size_t)BM * Cc;         // l_loss = l_fit + l_cut
  float* lossG = lossL + 1;                     // g_loss

  // capture-safe zeroing of accumulators
  hipMemsetAsync(W + O_ADJ, 0, (size_t)Bc * Mc * Mc * sizeof(float), stream);
  hipMemsetAsync(W + O_CTSUM, 0, ((size_t)Bc * NCLUSc * 3 + (size_t)Bc * NCLUSc) * sizeof(float), stream);
  hipMemsetAsync(lossL, 0, 2 * sizeof(float), stream);

  auto cdiv = [](size_t n, size_t d) { return (int)((n + d - 1) / d); };
  auto pack = [&](const float* src, size_t dst, int KdimSrc, int Kdim, int Ncols) {
    size_t total = pkSize(Kdim, Ncols);
    pack_b_kernel<<<cdiv(total, 256), 256, 0, stream>>>(src, HP + dst, KdimSrc, Kdim, Ncols);
  };

  // --- pack all B-operand weights to f16 fragment order ---
  pack(pos_w1, PK_POSW1, 72, PADF, 128);
  pack(pos_w2, PK_POSW2, 128, 128, Cc);
  pack(nei_w1, PK_NEIW1, 72, PADF, 128);
  pack(nei_w2, PK_NEIW2, 128, 128, Cc);
  pack(q_w, PK_QW, Cc, Cc, HIDc);
  pack(k_w, PK_KW, Cc, Cc, HIDc);
  pack(v_w, PK_VW, Cc, Cc, HIDc);
  pack(proj_w1, PK_PROJW1, Cc, Cc, HIDc);
  pack(proj_w2, PK_PROJW2, HIDc, HIDc, Cc);
  pack(cut_w1, PK_CUTW1, Cc, Cc, HIDc);
  pack(cut_w2, PK_CUTW2, HIDc, HIDc, NCLUSc);

  // --- positional MLPs ---
  pose_sp_kernel<<<cdiv((size_t)BM * PADF, 256), 256, 0, stream>>>(sp_xyz, W + O_POSE_S);
  gemm_wmma<EPI_GELU, 2><<<BM / 16, 128, 0, stream>>>(W + O_POSE_S, HP + PK_POSW1, pos_b1,
                                                      nullptr, nullptr, W + O_POSH, BM, PADF, 128);
  gemm_wmma<EPI_LN, 4><<<BM / 16, 128, 0, stream>>>(W + O_POSH, HP + PK_POSW2, pos_b2,
                                                    pos_ln_g, pos_ln_b, W + O_POSEMD, BM, 128, Cc);
  pose_rel_kernel<<<cdiv((size_t)BMK * PADF, 256), 256, 0, stream>>>(sp_xyz, p_xyz, topk, W + O_POSE_R);
  gemm_wmma<EPI_GELU, 2><<<BMK / 16, 128, 0, stream>>>(W + O_POSE_R, HP + PK_NEIW1, nei_b1,
                                                       nullptr, nullptr, W + O_NEIH, BMK, PADF, 128);
  gemm_wmma<EPI_LN, 4><<<BMK / 16, 128, 0, stream>>>(W + O_NEIH, HP + PK_NEIW2, nei_b2,
                                                     nei_ln_g, nei_ln_b, W + O_KV, BMK, 128, Cc);
  kv_add_kernel<<<cdiv((size_t)BMK * Cc, 256), 256, 0, stream>>>(p_fea, topk, W + O_KV);

  // --- attention ---
  add_kernel<<<cdiv((size_t)BM * Cc, 256), 256, 0, stream>>>(sp_fea, W + O_POSEMD, W + O_QIN,
                                                             (size_t)BM * Cc);
  gemm_wmma<EPI_BIAS, 4><<<BM / 16, 128, 0, stream>>>(W + O_QIN, HP + PK_QW, q_b,
                                                      nullptr, nullptr, W + O_Q, BM, Cc, HIDc);
  gemm_wmma<EPI_BIAS, 4><<<BMK / 16, 128, 0, stream>>>(W + O_KV, HP + PK_KW, k_b,
                                                       nullptr, nullptr, W + O_KF, BMK, Cc, HIDc);
  gemm_wmma<EPI_LN, 4><<<BMK / 16, 128, 0, stream>>>(W + O_KV, HP + PK_VW, v_b,
                                                     norm1_g, norm1_b, W + O_V, BMK, Cc, HIDc);
  attn_kernel<<<BM, 256, 0, stream>>>(W + O_Q, W + O_KF, W + O_V, mask, W + O_ATTN);
  norm_loss_kernel<<<BM, 256, 0, stream>>>(W + O_ATTN, sp_fea, lossL, 0.01f);   // l_fit

  // --- projection MLP with batchnorm ---
  add_ln_kernel<<<BM, 256, 0, stream>>>(W + O_ATTN, sp_fea, norm2_g, norm2_b, W + O_X);
  gemm_wmma<EPI_BIAS, 4><<<BM / 16, 128, 0, stream>>>(W + O_X, HP + PK_PROJW1, proj_b1,
                                                      nullptr, nullptr, W + O_HPRE, BM, Cc, HIDc);
  bn_stats_kernel<<<HIDc, 256, 0, stream>>>(W + O_HPRE, W + O_MEAN1, W + O_VAR1, BM, HIDc);
  bn_relu_kernel<<<cdiv((size_t)BM * HIDc, 256), 256, 0, stream>>>(W + O_HPRE, W + O_MEAN1, W + O_VAR1,
                                                                   proj_bn1_g, proj_bn1_b, W + O_HBN,
                                                                   (size_t)BM * HIDc, HIDc);
  gemm_wmma<EPI_NONE, 4><<<BM / 16, 128, 0, stream>>>(W + O_HBN, HP + PK_PROJW2, nullptr,
                                                      nullptr, nullptr, W + O_FPRE, BM, HIDc, Cc);
  bn_stats_kernel<<<Cc, 256, 0, stream>>>(W + O_FPRE, W + O_MEAN2, W + O_VAR2, BM, Cc);
  bn_relu_kernel<<<cdiv((size_t)BM * Cc, 256), 256, 0, stream>>>(W + O_FPRE, W + O_MEAN2, W + O_VAR2,
                                                                 proj_bn2_g, proj_bn2_b, spf_new,
                                                                 (size_t)BM * Cc, Cc);
  norm_loss_kernel<<<BM, 256, 0, stream>>>(spf_new, sp_fea, lossG, 1.0f);       // g_loss

  // --- min-cut branch ---
  knn_kernel<<<BM / 128, 128, 0, stream>>>(sp_xyz, W + O_ADJ);
  degree_kernel<<<BM, 256, 0, stream>>>(W + O_ADJ, W + O_DINV);
  scale_adj_kernel<<<cdiv((size_t)Bc * Mc * Mc, 256), 256, 0, stream>>>(W + O_ADJ, W + O_DINV);
  for (int b = 0; b < Bc; ++b) {                 // pack sp_fea_new as B, then hg = A' @ spf
    pack(spf_new + (size_t)b * Mc * Cc, PK_SPF + (size_t)b * pkSize(Mc, Cc), Mc, Mc, Cc);
    gemm_wmma<EPI_NONE, 4><<<Mc / 16, 128, 0, stream>>>(W + O_ADJ + (size_t)b * Mc * Mc,
                                                        HP + PK_SPF + (size_t)b * pkSize(Mc, Cc),
                                                        nullptr, nullptr, nullptr,
                                                        W + O_HG + (size_t)b * Mc * Cc, Mc, Mc, Cc);
  }
  gemm_wmma<EPI_RELU, 4><<<BM / 16, 128, 0, stream>>>(W + O_HG, HP + PK_CUTW1, cut_b1,
                                                      nullptr, nullptr, W + O_CUT1, BM, Cc, HIDc);
  gemm_wmma<EPI_BIAS, 1><<<BM / 16, 128, 0, stream>>>(W + O_CUT1, HP + PK_CUTW2, cut_b2,
                                                      nullptr, nullptr, W + O_SLOG, BM, HIDc, NCLUSc);
  cut_softmax_kernel<<<BM, 64, 0, stream>>>(W + O_SLOG, W + O_S);
  cts_accum_kernel<<<BM / 256, 256, 0, stream>>>(W + O_S, sp_xyz, W + O_CTSUM, W + O_SSUM);
  cut_loss_kernel<<<BM / 256, 256, 0, stream>>>(W + O_S, sp_xyz, W + O_CTSUM, W + O_SSUM, lossL); // l_cut
}
